// Condition_LSTM_5772436046454
// MI455X (gfx1250) — matrix operand order repaired and checked by reading, not compiled, not run on hardware
//
#include <hip/hip_runtime.h>
#include <hip/hip_bf16.h>
#include <stdint.h>

// ---------------------------------------------------------------------------
// CDNA5 (gfx1250) implementation of the Graves attention-LSTM reference.
// bf16 WMMA (v_wmma_f32_16x16x32_bf16) for all GEMMs, fp32 accumulate.
// ---------------------------------------------------------------------------

typedef __attribute__((ext_vector_type(16))) __bf16 v16bf;
typedef __attribute__((ext_vector_type(8)))  float  v8f;

namespace cfg {
constexpr int B  = 128, T = 512, IN = 3, A = 77, H = 512;
constexpr int KW = 10, KM = 20, U = 64;
constexpr int OUT = 6 * KM + 1;          // 121
constexpr int K1IN = IN + A + H;         // 592  (x, w, h1_prev)
constexpr int K1PAD = 608, KT1 = 19;
constexpr int K2IN = IN + A + H + H;     // 1104 (x, w, h1, h2_prev)
constexpr int K2PAD = 1120, KT2 = 35;
constexpr int KTF = 16;                  // fc2: K = 512
constexpr int NT1 = 128;                 // 2048 / 16 N-tiles for gate GEMMs
constexpr int NTF = 8;                   // 128 / 16 (121 padded to 128)
}

#define DEVFN static __device__ __forceinline__

DEVFN float sigm(float x) { return 1.f / (1.f + __expf(-x)); }

DEVFN uint16_t f32_to_bf16(float f) {
  uint32_t u = __builtin_bit_cast(uint32_t, f);
  uint32_t r = u + 0x7FFFu + ((u >> 16) & 1u);   // round-to-nearest-even
  return (uint16_t)(r >> 16);
}

struct Dw8 { uint32_t d[8]; };

// A fragment (16x32 bf16, MxK), ISA 7.12.2 layout:
//   lane 0-15 : M=lane,    VGPR0-3 K=0..7,  VGPR4-7 K=16..23
//   lane16-31 : M=lane-16, VGPR0-3 K=8..15, VGPR4-7 K=24..31
DEVFN v16bf load_a_frag(const uint16_t* base, int rowstride, int m0, int kt, int lane) {
  const int m  = m0 + (lane & 15);
  const int kb = (lane < 16) ? 0 : 8;
  const uint16_t* row = base + (size_t)m * rowstride + kt * 32;
  Dw8 f;
#pragma unroll
  for (int v = 0; v < 8; ++v) {
    const int k = (v < 4) ? (2 * v + kb) : (16 + 2 * (v - 4) + kb);
    f.d[v] = *reinterpret_cast<const uint32_t*>(row + k);
  }
  return __builtin_bit_cast(v16bf, f);
}

// B fragment (32x16 bf16, KxN) pre-packed: 1024B tile, lane reads 32 bytes.
DEVFN v16bf load_b_frag(const uint32_t* tile, int lane) {
  const uint4* p = reinterpret_cast<const uint4*>(tile + lane * 8);
  uint4 a = p[0], b = p[1];
  Dw8 f;
  f.d[0]=a.x; f.d[1]=a.y; f.d[2]=a.z; f.d[3]=a.w;
  f.d[4]=b.x; f.d[5]=b.y; f.d[6]=b.z; f.d[7]=b.w;
  return __builtin_bit_cast(v16bf, f);
}

// ---------------------------------------------------------------------------
// Weight pre-pack: B[k][n] = Wcat[n][k], Wcat[n] = [src1_row(n) | src2_row(n) | 0-pad]
// Tile (nt,kt) at dword offset (nt*KT+kt)*256; dword r: lane=r>>3, v=r&7;
// n = nt*16 + (lane&15); k = kt*32 + ((lane<16)?0:8) + (v<4 ? 2v : 16+2(v-4)); packs bf16(k),bf16(k+1).
// ---------------------------------------------------------------------------
__global__ void pack_weights_kernel(const float* __restrict__ src1, int K1,
                                    const float* __restrict__ src2, int K2,
                                    int Nact, int Ntiles, int KT,
                                    uint32_t* __restrict__ dst) {
  const size_t idx = (size_t)blockIdx.x * 256 + threadIdx.x;
  const size_t total = (size_t)Ntiles * KT * 256;
  if (idx >= total) return;
  const int r = (int)(idx & 255);
  const size_t tile = idx >> 8;
  const int kt = (int)(tile % KT);
  const int nt = (int)(tile / KT);
  const int lane = r >> 3, v = r & 7;
  const int n = nt * 16 + (lane & 15);
  const int k = kt * 32 + ((lane < 16) ? 0 : 8) + ((v < 4) ? 2 * v : 16 + 2 * (v - 4));
  const int Ktot = K1 + K2;
  uint32_t out = 0;
  if (n < Nact) {
    float e0 = 0.f, e1 = 0.f;
    if (k < K1)            e0 = src1[(size_t)n * K1 + k];
    else if (k < Ktot)     e0 = src2[(size_t)n * K2 + (k - K1)];
    if (k + 1 < K1)        e1 = src1[(size_t)n * K1 + k + 1];
    else if (k + 1 < Ktot) e1 = src2[(size_t)n * K2 + (k + 1 - K1)];
    out = (uint32_t)f32_to_bf16(e0) | ((uint32_t)f32_to_bf16(e1) << 16);
  }
  dst[idx] = out;
}

__global__ void combine_bias_kernel(const float* __restrict__ a,
                                    const float* __restrict__ b,
                                    float* __restrict__ dst, int n) {
  int i = blockIdx.x * 256 + threadIdx.x;
  if (i < n) dst[i] = a[i] + b[i];
}

__global__ void zero_dwords_kernel(uint32_t* __restrict__ p, size_t n) {
  size_t stride = (size_t)gridDim.x * blockDim.x;
  for (size_t i = (size_t)blockIdx.x * blockDim.x + threadIdx.x; i < n; i += stride)
    p[i] = 0u;
}

__global__ void write_x0_kernel(uint16_t* __restrict__ A1cur, const float* __restrict__ x) {
  int i = blockIdx.x * 64 + threadIdx.x;               // B*IN = 384
  if (i < cfg::B * cfg::IN) {
    int b = i / cfg::IN, k = i % cfg::IN;
    A1cur[(size_t)b * cfg::K1PAD + k] = f32_to_bf16(x[(size_t)b * cfg::T * cfg::IN + k]);
  }
}

// ---------------------------------------------------------------------------
// Fused gate-GEMM (M=128, N=2048) + LSTM cell update.
// grid (16, 8): x = 32-hidden-col group, y = 16-row batch tile. 256 thr = 8 waves.
// Wave w: gate = w>>1, half = w&1 -> N-tile gate*32 + bx*2 + half, so the block
// owns all four gates for its 16x32 output slab and can apply the cell in LDS.
// ---------------------------------------------------------------------------
__global__ __launch_bounds__(256) void lstm_step_kernel(
    const uint16_t* __restrict__ Ain, int rowstride, int KT,
    const uint32_t* __restrict__ Wpack,
    const float*   __restrict__ bias,
    float* __restrict__ cbuf, float* __restrict__ hbuf,
    uint16_t* __restrict__ hdst0, int hstride0,
    uint16_t* __restrict__ hdst1, int hstride1) {
  __shared__ float zbuf[8][16][16];
  const int tid = threadIdx.x, lane = tid & 31, w = tid >> 5;
  const int gate = w >> 1, jh = w & 1;
  const int j0 = blockIdx.x * 32, m0 = blockIdx.y * 16;
  const int ntile = gate * 32 + (blockIdx.x * 2 + jh);

  v8f acc = {0.f, 0.f, 0.f, 0.f, 0.f, 0.f, 0.f, 0.f};
  const uint32_t* wt = Wpack + (size_t)ntile * KT * 256;
  for (int kt = 0; kt < KT; ++kt) {
    v16bf a = load_a_frag(Ain, rowstride, m0, kt, lane);
    v16bf b = load_b_frag(wt + (size_t)kt * 256, lane);
    acc = __builtin_amdgcn_wmma_f32_16x16x32_bf16(false, a, false, b, (short)0,
                                                  acc, false, false);
  }
  const float bv = bias[ntile * 16 + (lane & 15)];
  const int mb = (lane < 16) ? 0 : 8, nc = lane & 15;
#pragma unroll
  for (int r = 0; r < 8; ++r) zbuf[w][mb + r][nc] = acc[r] + bv;
  __syncthreads();

  for (int e = tid; e < 512; e += 256) {
    const int m = e >> 5, jj = e & 31, hcol = jj >> 4, c16 = jj & 15;
    const float zi = zbuf[0 + hcol][m][c16];
    const float zf = zbuf[2 + hcol][m][c16];
    const float zg = zbuf[4 + hcol][m][c16];
    const float zo = zbuf[6 + hcol][m][c16];
    const int b = m0 + m, j = j0 + jj;
    const size_t idx = (size_t)b * cfg::H + j;
    const float cprev = cbuf[idx];
    const float cn = sigm(zf) * cprev + sigm(zi) * tanhf(zg);
    const float hn = sigm(zo) * tanhf(cn);
    cbuf[idx] = cn;
    hbuf[idx] = hn;
    const uint16_t hb = f32_to_bf16(hn);
    hdst0[(size_t)b * hstride0 + j] = hb;
    if (hdst1) hdst1[(size_t)b * hstride1 + j] = hb;
  }
}

// ---------------------------------------------------------------------------
// Graves attention window; one block per batch row.
// Also packs bf16 activations: w_t + x_t -> A2(cur); w_t + x_{t+1} -> A1(next).
// ---------------------------------------------------------------------------
__global__ __launch_bounds__(256) void attn_kernel(
    const float* __restrict__ h1, const float* __restrict__ Wabk,
    const float* __restrict__ babk, float* __restrict__ kappa_ws,
    const float* __restrict__ x, const float* __restrict__ c_seq,
    uint16_t* __restrict__ A2cur, uint16_t* __restrict__ A1next, int t) {
  const int b = blockIdx.x, tid = threadIdx.x;
  __shared__ float red[30][8];
  __shared__ float abk[30];
  __shared__ float alpha[cfg::KW], beta[cfg::KW], kap[cfg::KW];
  __shared__ float phi[cfg::U];

  if (tid < 240) {                       // 30 dot products of length 512
    const int r = tid >> 3, g = tid & 7;
    const float* hr = h1 + (size_t)b * cfg::H;
    const float* wr = Wabk + (size_t)r * cfg::H;
    float s = 0.f;
    for (int k = g * 64; k < g * 64 + 64; ++k) s += hr[k] * wr[k];
    red[r][g] = s;
  }
  __syncthreads();
  if (tid < 30) {
    float s = babk[tid];
#pragma unroll
    for (int g = 0; g < 8; ++g) s += red[tid][g];
    abk[tid] = s;
  }
  __syncthreads();
  if (tid < cfg::KW) {
    alpha[tid] = __expf(abk[tid]);
    beta[tid]  = __expf(abk[cfg::KW + tid]);
    float k = kappa_ws[b * cfg::KW + tid] + __expf(abk[2 * cfg::KW + tid]);
    kappa_ws[b * cfg::KW + tid] = k;
    kap[tid] = k;
  }
  __syncthreads();
  if (tid < cfg::U) {
    const float u = (float)tid;
    float s = 0.f;
#pragma unroll
    for (int k = 0; k < cfg::KW; ++k) {
      const float d = kap[k] - u;
      s += alpha[k] * __expf(-beta[k] * d * d);
    }
    phi[tid] = s;
  }
  __syncthreads();
  if (tid < cfg::A) {                    // w_t[a] = sum_u phi[u] * c_seq[b,u,a]
    const float* cs = c_seq + (size_t)b * cfg::U * cfg::A + tid;
    float s = 0.f;
    for (int u = 0; u < cfg::U; ++u) s += phi[u] * cs[(size_t)u * cfg::A];
    const uint16_t wv = f32_to_bf16(s);
    A2cur [(size_t)b * cfg::K2PAD + cfg::IN + tid] = wv;
    A1next[(size_t)b * cfg::K1PAD + cfg::IN + tid] = wv;
  }
  if (tid < cfg::IN) {
    A2cur[(size_t)b * cfg::K2PAD + tid] =
        f32_to_bf16(x[((size_t)b * cfg::T + t) * cfg::IN + tid]);
    if (t + 1 < cfg::T)
      A1next[(size_t)b * cfg::K1PAD + tid] =
          f32_to_bf16(x[((size_t)b * cfg::T + t + 1) * cfg::IN + tid]);
  }
}

// ---------------------------------------------------------------------------
// fc2: out = h2 @ W_fc2^T + b.  M=128, N=121 (padded 128), K=512.
// grid 8 (M-tiles), 8 waves = 8 N-tiles. Reads bf16 h2 slab inside A2(next).
// ---------------------------------------------------------------------------
__global__ __launch_bounds__(256) void fc2_kernel(
    const uint16_t* __restrict__ Ah2, int rowstride,
    const uint32_t* __restrict__ Wpack, const float* __restrict__ bias,
    float* __restrict__ outs, int t) {
  const int tid = threadIdx.x, lane = tid & 31, w = tid >> 5;
  const int m0 = blockIdx.x * 16;
  v8f acc = {0.f, 0.f, 0.f, 0.f, 0.f, 0.f, 0.f, 0.f};
  const uint32_t* wt = Wpack + (size_t)w * cfg::KTF * 256;
  for (int kt = 0; kt < cfg::KTF; ++kt) {
    v16bf a = load_a_frag(Ah2, rowstride, m0, kt, lane);
    v16bf b = load_b_frag(wt + (size_t)kt * 256, lane);
    acc = __builtin_amdgcn_wmma_f32_16x16x32_bf16(false, a, false, b, (short)0,
                                                  acc, false, false);
  }
  const int n = w * 16 + (lane & 15);
  if (n < cfg::OUT) {
    const float bv = bias[n];
    const int mb = (lane < 16) ? 0 : 8;
#pragma unroll
    for (int r = 0; r < 8; ++r) {
      const int b = m0 + mb + r;
      outs[((size_t)b * cfg::T + t) * cfg::OUT + n] = acc[r] + bv;
    }
  }
}

// ---------------------------------------------------------------------------
// MDN head post-processing; one block per (b,t).
// d_out layout: [pi|mu1|mu2|std1|std2|corr] each B*T*20, then eot B*T.
// ---------------------------------------------------------------------------
__global__ __launch_bounds__(128) void mdn_post_kernel(const float* __restrict__ outs,
                                                       float* __restrict__ out) {
  const size_t bt = blockIdx.x;
  const float* z = outs + bt * cfg::OUT;
  __shared__ float sm[2];
  const int tid = threadIdx.x;
  if (tid == 0) {
    float mx = z[0];
    for (int k = 1; k < cfg::KM; ++k) mx = fmaxf(mx, z[k]);
    float s = 0.f;
    for (int k = 0; k < cfg::KM; ++k) s += __expf(z[k] - mx);
    sm[0] = mx; sm[1] = s;
  }
  __syncthreads();
  const size_t S = (size_t)cfg::B * cfg::T * cfg::KM;
  if (tid < 20)        out[0 * S + bt * 20 + tid]         = __expf(z[tid] - sm[0]) / sm[1];
  else if (tid < 40)   out[1 * S + bt * 20 + (tid - 20)]  = z[tid];
  else if (tid < 60)   out[2 * S + bt * 20 + (tid - 40)]  = z[tid];
  else if (tid < 80)   out[3 * S + bt * 20 + (tid - 60)]  = __expf(z[tid]);
  else if (tid < 100)  out[4 * S + bt * 20 + (tid - 80)]  = __expf(z[tid]);
  else if (tid < 120)  out[5 * S + bt * 20 + (tid - 100)] = tanhf(z[tid]);
  else if (tid == 120) out[6 * S + bt]                    = sigm(z[120]);
}

// ---------------------------------------------------------------------------
extern "C" void kernel_launch(void* const* d_in, const int* in_sizes, int n_in,
                              void* d_out, int out_size, void* d_ws, size_t ws_size,
                              hipStream_t stream) {
  using namespace cfg;
  const float* x     = (const float*)d_in[0];
  const float* c_seq = (const float*)d_in[1];
  const float* Wih1  = (const float*)d_in[2];
  const float* Whh1  = (const float*)d_in[3];
  const float* bih1  = (const float*)d_in[4];
  const float* bhh1  = (const float*)d_in[5];
  const float* Wih2  = (const float*)d_in[6];
  const float* Whh2  = (const float*)d_in[7];
  const float* bih2  = (const float*)d_in[8];
  const float* bhh2  = (const float*)d_in[9];
  const float* Wabk  = (const float*)d_in[10];
  const float* babk  = (const float*)d_in[11];
  const float* Wfc2  = (const float*)d_in[12];
  const float* bfc2  = (const float*)d_in[13];
  (void)in_sizes; (void)n_in; (void)out_size; (void)ws_size;

  // --- workspace carve-up (all chunks 256B aligned) ---
  char* ws = (char*)d_ws;
  size_t off = 0;
  auto carve = [&](size_t bytes) -> char* {
    char* p = ws + off;
    off += (bytes + 255) & ~(size_t)255;
    return p;
  };
  uint32_t* w1pack = (uint32_t*)carve((size_t)NT1 * KT1 * 1024);  // 2.49 MB
  uint32_t* w2pack = (uint32_t*)carve((size_t)NT1 * KT2 * 1024);  // 4.59 MB
  uint32_t* wfpack = (uint32_t*)carve((size_t)NTF * KTF * 1024);  // 128 KB
  float*    bc1    = (float*)carve(4 * H * sizeof(float));
  float*    bc2    = (float*)carve(4 * H * sizeof(float));
  // contiguous per-call state block (zeroed every call)
  char* state0 = ws + off;
  uint16_t* A1buf = (uint16_t*)carve((size_t)2 * B * K1PAD * 2);  // double-buffered
  uint16_t* A2buf = (uint16_t*)carve((size_t)2 * B * K2PAD * 2);
  float*    c1ws  = (float*)carve((size_t)B * H * 4);
  float*    c2ws  = (float*)carve((size_t)B * H * 4);
  float*    kapws = (float*)carve((size_t)B * KW * 4);
  size_t state_dwords = (size_t)((ws + off) - state0) / 4;
  float* h1ws  = (float*)carve((size_t)B * H * 4);
  float* h2ws  = (float*)carve((size_t)B * H * 4);
  float* outsw = (float*)carve((size_t)B * T * OUT * 4);          // 31.7 MB

  // --- per-call preprocessing ---
  {
    size_t t1 = (size_t)NT1 * KT1 * 256;
    pack_weights_kernel<<<(unsigned)((t1 + 255) / 256), 256, 0, stream>>>(
        Wih1, IN + A, Whh1, H, 4 * H, NT1, KT1, w1pack);
    size_t t2 = (size_t)NT1 * KT2 * 256;
    pack_weights_kernel<<<(unsigned)((t2 + 255) / 256), 256, 0, stream>>>(
        Wih2, IN + A + H, Whh2, H, 4 * H, NT1, KT2, w2pack);
    size_t tf = (size_t)NTF * KTF * 256;
    pack_weights_kernel<<<(unsigned)((tf + 255) / 256), 256, 0, stream>>>(
        Wfc2, H, nullptr, 0, OUT, NTF, KTF, wfpack);
    combine_bias_kernel<<<(4 * H + 255) / 256, 256, 0, stream>>>(bih1, bhh1, bc1, 4 * H);
    combine_bias_kernel<<<(4 * H + 255) / 256, 256, 0, stream>>>(bih2, bhh2, bc2, 4 * H);
    zero_dwords_kernel<<<512, 256, 0, stream>>>((uint32_t*)state0, state_dwords);
    write_x0_kernel<<<(B * IN + 63) / 64, 64, 0, stream>>>(A1buf, x);
  }

  // --- sequential scan over T timesteps ---
  const dim3 gGate(16, 8);   // 16 hidden-col groups of 32, 8 batch tiles of 16
  for (int t = 0; t < T; ++t) {
    uint16_t* A1cur = A1buf + (size_t)(t & 1) * B * K1PAD;
    uint16_t* A1nxt = A1buf + (size_t)((t + 1) & 1) * B * K1PAD;
    uint16_t* A2cur = A2buf + (size_t)(t & 1) * B * K2PAD;
    uint16_t* A2nxt = A2buf + (size_t)((t + 1) & 1) * B * K2PAD;

    // LSTM1: reads A1(cur); writes h1 -> A2(cur)[80..591] and A1(next)[80..591]
    lstm_step_kernel<<<gGate, 256, 0, stream>>>(
        A1cur, K1PAD, KT1, w1pack, bc1, c1ws, h1ws,
        A2cur + (IN + A), K2PAD, A1nxt + (IN + A), K1PAD);

    // attention: writes w_t,x -> A2(cur)[0..79] and A1(next)[0..79]
    attn_kernel<<<B, 256, 0, stream>>>(h1ws, Wabk, babk, kapws, x, c_seq,
                                       A2cur, A1nxt, t);

    // LSTM2: reads A2(cur); writes h2 -> A2(next)[592..1103]
    lstm_step_kernel<<<gGate, 256, 0, stream>>>(
        A2cur, K2PAD, KT2, w2pack, bc2, c2ws, h2ws,
        A2nxt + (IN + A + H), K2PAD, nullptr, 0);

    // fc2 on the freshly written bf16 h2 slab
    fc2_kernel<<<NTF, 256, 0, stream>>>(A2nxt + (IN + A + H), K2PAD,
                                        wfpack, bfc2, outsw, t);
  }

  // --- MDN head post-processing for all (b,t) ---
  mdn_post_kernel<<<B * T, 128, 0, stream>>>(outsw, (float*)d_out);
}
